// MultiHeadAttention_88278757802278
// MI455X (gfx1250) — matrix operand order repaired and checked by reading, not compiled, not run on hardware
//
#include <hip/hip_runtime.h>

typedef unsigned short u16;
typedef __attribute__((ext_vector_type(16))) __bf16 v16bf;
typedef __attribute__((ext_vector_type(8)))  float  v8f;
typedef __attribute__((ext_vector_type(4)))  unsigned int v4u;
typedef __attribute__((ext_vector_type(4)))  int v4i;

#define B_    4
#define S_    2048
#define D_    1024
#define NH_   16
#define DK_   64
#define BS_   (B_ * S_)   // 8192

// GEMM block tiling
#define TM_   128
#define TN_   64
#define TK_   32

union FragU { v4u q[2]; v16bf v; };

__device__ __forceinline__ v16bf load_f2(const u16* p0, const u16* p1) {
  FragU u;
  u.q[0] = *(const v4u*)p0;   // 8 bf16 (16B)
  u.q[1] = *(const v4u*)p1;   // 8 bf16 (16B)
  return u.v;
}
__device__ __forceinline__ v16bf load_f1(const u16* p) { return load_f2(p, p + 8); }

__device__ __forceinline__ u16 f2bf(float f) {
  unsigned int u = __float_as_uint(f);
  u += 0x7fffu + ((u >> 16) & 1u);   // round-to-nearest-even
  return (u16)(u >> 16);
}

__device__ __forceinline__ v8f wmma_bf16(v16bf a, v16bf b, v8f c) {
  // (neg_a, A, neg_b, B, c_mod, C, reuse_a, reuse_b) — probe-confirmed 8-arg form
  return __builtin_amdgcn_wmma_f32_16x16x32_bf16(false, a, false, b, (short)0, c, false, false);
}

// ---- CDNA5 async global->LDS copy (ASYNCcnt path), guarded with fallback ----
__device__ __forceinline__ void async_copy16(const u16* g, u16* l) {
#if __has_builtin(__builtin_amdgcn_global_load_async_to_lds_b128)
  typedef __attribute__((address_space(1))) v4i* gp_t;   // global v4i*
  typedef __attribute__((address_space(3))) v4i* lp_t;   // LDS v4i*
  __builtin_amdgcn_global_load_async_to_lds_b128(
      (gp_t)(unsigned long long)(size_t)g,
      (lp_t)(unsigned int)(size_t)l,          // low 32 bits of flat LDS addr = LDS offset
      0 /*imm offset*/, 0 /*cpol*/);
#else
  *(v4u*)l = *(const v4u*)g;
#endif
}

template <int N>
__device__ __forceinline__ void wait_async() {
#if __has_builtin(__builtin_amdgcn_s_wait_asynccnt)
  __builtin_amdgcn_s_wait_asynccnt((unsigned short)N);
#else
  asm volatile("s_wait_asynccnt %0" :: "i"(N) : "memory");
#endif
}

// ---------------- f32 -> bf16 cast ----------------
__global__ void __launch_bounds__(256) mha_cast_bf16(const float* __restrict__ in,
                                                     u16* __restrict__ out, int n) {
  int i = blockIdx.x * blockDim.x + threadIdx.x;
  int stride = gridDim.x * blockDim.x;
  for (; i < n; i += stride) out[i] = f2bf(in[i]);
}

// ---------------- C[M,N] = A[M,K] * W[N,K]^T  (bf16 in, f32 out) ----------------
// Cooperative: block (8 waves) stages 128x32 A-tile + 64x32 B-tile in LDS via
// async copies (double-buffered); each wave computes a 32x32 tile with 2x2 WMMA frags.
__global__ void __launch_bounds__(256) mha_gemm_bf16(const u16* __restrict__ A,
                                                     const u16* __restrict__ W,
                                                     float* __restrict__ C,
                                                     int M, int N, int K) {
  __shared__ alignas(16) u16 At[2][TM_ * TK_];   // 2 x 8KB
  __shared__ alignas(16) u16 Bt[2][TN_ * TK_];   // 2 x 4KB

  const int tid  = threadIdx.x;
  const int lane = tid & 31;
  const int wv   = tid >> 5;                     // 0..7
  const int hh   = lane >> 4, lr = lane & 15;

  const int ntn = N / TN_;
  const int m0  = (blockIdx.x / ntn) * TM_;
  const int n0  = (blockIdx.x % ntn) * TN_;
  const int wm  = (wv >> 1) * 32;                // wave sub-tile: 4 x 2 arrangement
  const int wn  = (wv & 1) * 32;

  // per-step copy: A = 512 16B-chunks (2/thread), B = 256 16B-chunks (1/thread)
  auto issue_copy = [&](int buf, int kb) {
#pragma unroll
    for (int cc = 0; cc < 2; ++cc) {
      int c   = tid + cc * 256;
      int row = c >> 2, kc = (c & 3) * 8;
      async_copy16(A + (size_t)(m0 + row) * K + kb + kc, &At[buf][row * TK_ + kc]);
    }
    {
      int row = tid >> 2, kc = (tid & 3) * 8;
      async_copy16(W + (size_t)(n0 + row) * K + kb + kc, &Bt[buf][row * TK_ + kc]);
    }
  };

  v8f acc[2][2] = {};
  const int nk = K / TK_;
  issue_copy(0, 0);
  for (int kbi = 0; kbi < nk; ++kbi) {
    const int buf = kbi & 1;
    if (kbi + 1 < nk) {
      issue_copy(buf ^ 1, (kbi + 1) * TK_);      // prefetch next tile
      wait_async<3>();                            // only the 3 newest may remain
    } else {
      wait_async<0>();
    }
    __syncthreads();                              // all waves' copies visible

    v16bf af[2], bfr[2];
#pragma unroll
    for (int i = 0; i < 2; ++i) {
      const u16* pa = &At[buf][(wm + 16 * i + lr) * TK_ + hh * 8];
      af[i] = load_f2(pa, pa + 16);               // A: K = {h*8+0..7, 16+h*8+0..7}
      const u16* pb = &Bt[buf][(wn + 16 * i + lr) * TK_ + hh * 8];
      bfr[i] = load_f2(pb, pb + 16);              // B: K = {h*8+0..7, 16+h*8+0..7}
    }
#pragma unroll
    for (int i = 0; i < 2; ++i)
#pragma unroll
      for (int j = 0; j < 2; ++j)
        acc[i][j] = wmma_bf16(af[i], bfr[j], acc[i][j]);

    __syncthreads();                              // done reading buf before reuse
  }

#pragma unroll
  for (int i = 0; i < 2; ++i)
#pragma unroll
    for (int j = 0; j < 2; ++j)
#pragma unroll
      for (int r = 0; r < 8; ++r) {
        int row = m0 + wm + 16 * i + hh * 8 + r;  // C/D layout: M = h*8 + vgpr_idx
        int col = n0 + wn + 16 * j + lr;          // N = lane&15
        C[(size_t)row * N + col] = acc[i][j][r];
      }
}

// ---------------- RoPE + head-split: raw f32 [BS,D] -> bf16 [B,NH,S,DK] ----------------
__global__ void __launch_bounds__(256) mha_rope_split(const float* __restrict__ raw,
                                                      u16* __restrict__ out, int total) {
  int idx = blockIdx.x * blockDim.x + threadIdx.x;
  if (idx >= total) return;                 // total = BS * (D/2) pairs
  int dp = idx & 511;                       // pair index in d_model
  int sg = idx >> 9;                        // token index in [0, BS)
  int b = sg >> 11, s = sg & 2047;
  int d = dp << 1;
  int h = d >> 6, dd = d & 63;              // dd even, position in head
  float inv_freq = __powf(10000.0f, -(float)dd * (1.0f / 64.0f));
  float ang = (float)s * inv_freq;
  float sn, cs;
  __sincosf(ang, &sn, &cs);
  float x1 = raw[(size_t)sg * D_ + d];
  float x2 = raw[(size_t)sg * D_ + d + 1];
  size_t base = (((size_t)(b * NH_ + h) * S_) + s) * DK_ + dd;
  out[base]     = f2bf(x1 * cs - x2 * sn);
  out[base + 1] = f2bf(x1 * sn + x2 * cs);
}

// ---------------- V split + transpose: raw f32 [BS,D] -> bf16 [B,NH,DK,S] ----------------
__global__ void __launch_bounds__(256) mha_v_split(const float* __restrict__ raw,
                                                   u16* __restrict__ vt, int total) {
  int idx = blockIdx.x * blockDim.x + threadIdx.x;
  if (idx >= total) return;                 // total = BS * D
  int dmi = idx & 1023; int sg = idx >> 10;
  int b = sg >> 11, s = sg & 2047;
  int h = dmi >> 6, dd = dmi & 63;
  vt[(((size_t)(b * NH_ + h) * DK_ + dd) * S_) + s] = f2bf(raw[(size_t)sg * D_ + dmi]);
}

// ---------------- causal flash attention: one wave per (b,h,16-query tile) ----------------
__global__ void __launch_bounds__(256) mha_attention(const u16* __restrict__ Q,
                                                     const u16* __restrict__ Km,
                                                     const u16* __restrict__ Vt,
                                                     u16* __restrict__ Ctx) {
  __shared__ alignas(16) u16 Plds[8][16 * 32];  // per-wave P staging (C-layout -> A-layout)
  const int lane = threadIdx.x & 31;
  const int wib  = threadIdx.x >> 5;
  const int wid  = blockIdx.x * 8 + wib;    // [0, 8192)
  const int qt = wid & 127;
  const int h  = (wid >> 7) & 15;
  const int b  = wid >> 11;
  const int q0 = qt * 16;
  const int hh = lane >> 4, lr = lane & 15;

  const size_t headOff = (size_t)(b * NH_ + h) * S_ * DK_;
  const u16* Qh = Q  + headOff;             // [S, DK]
  const u16* Kh = Km + headOff;             // [S, DK]
  const u16* Vh = Vt + headOff;             // [DK, S]

  v16bf aq[2];
  {
    const u16* p0 = Qh + (size_t)(q0 + lr) * DK_ + 0  + hh * 8;
    aq[0] = load_f2(p0, p0 + 16);
    const u16* p1 = Qh + (size_t)(q0 + lr) * DK_ + 32 + hh * 8;
    aq[1] = load_f2(p1, p1 + 16);
  }

  float mrun[8], lrun[8];
#pragma unroll
  for (int r = 0; r < 8; ++r) { mrun[r] = -1e30f; lrun[r] = 0.0f; }
  v8f o[4] = {};

  const int kend = q0 + 16;                 // causal: keys <= q0+15 only
  for (int kb = 0; kb < kend; kb += 32) {
    v8f sc[2];
#pragma unroll
    for (int t = 0; t < 2; ++t) {
      const u16* pk = Kh + (size_t)(kb + 16 * t + lr) * DK_;
      v16bf bk0 = load_f2(pk + 0  + hh * 8, pk + 16 + hh * 8);
      v16bf bk1 = load_f2(pk + 32 + hh * 8, pk + 48 + hh * 8);
      v8f a = {};
      a = wmma_bf16(aq[0], bk0, a);
      a = wmma_bf16(aq[1], bk1, a);
      sc[t] = a;
    }
    float p[2][8], vm[8];
#pragma unroll
    for (int r = 0; r < 8; ++r) {
      int row = q0 + hh * 8 + r;
#pragma unroll
      for (int t = 0; t < 2; ++t) {
        int key = kb + 16 * t + lr;
        float s = sc[t][r] * 0.125f;        // 1/sqrt(64)
        p[t][r] = (key <= row) ? s : -1e30f;
      }
      vm[r] = fmaxf(p[0][r], p[1][r]);
#pragma unroll
      for (int off = 1; off < 16; off <<= 1)
        vm[r] = fmaxf(vm[r], __shfl_xor(vm[r], off, 32));
    }
#pragma unroll
    for (int r = 0; r < 8; ++r) {
      float mnew  = fmaxf(mrun[r], vm[r]);
      float scale = __expf(mrun[r] - mnew);
      p[0][r] = __expf(p[0][r] - mnew);
      p[1][r] = __expf(p[1][r] - mnew);
      float ps = p[0][r] + p[1][r];
#pragma unroll
      for (int off = 1; off < 16; off <<= 1)
        ps += __shfl_xor(ps, off, 32);
      lrun[r] = lrun[r] * scale + ps;
      mrun[r] = mnew;
#pragma unroll
      for (int f = 0; f < 4; ++f) o[f][r] *= scale;
    }
#pragma unroll
    for (int t = 0; t < 2; ++t)
#pragma unroll
      for (int r = 0; r < 8; ++r)
        Plds[wib][(hh * 8 + r) * 32 + 16 * t + lr] = f2bf(p[t][r]);
    v16bf ap;
    {
      const u16* pp = &Plds[wib][lr * 32 + hh * 8];
      ap = load_f2(pp, pp + 16);
    }
#pragma unroll
    for (int f = 0; f < 4; ++f) {
      const u16* pv = Vh + (size_t)(16 * f + lr) * S_ + kb + hh * 8;
      v16bf bv = load_f2(pv, pv + 16);
      o[f] = wmma_bf16(ap, bv, o[f]);
    }
  }
#pragma unroll
  for (int r = 0; r < 8; ++r) {
    float inv = 1.0f / lrun[r];
    int row = q0 + hh * 8 + r;
#pragma unroll
    for (int f = 0; f < 4; ++f) {
      int col = h * DK_ + 16 * f + lr;
      Ctx[((size_t)(b * S_ + row)) * D_ + col] = f2bf(o[f][r] * inv);
    }
  }
}

extern "C" void kernel_launch(void* const* d_in, const int* in_sizes, int n_in,
                              void* d_out, int out_size, void* d_ws, size_t ws_size,
                              hipStream_t stream) {
  const float* x  = (const float*)d_in[0];
  const float* wq = (const float*)d_in[1];
  const float* wk = (const float*)d_in[2];
  const float* wv = (const float*)d_in[3];
  const float* wo = (const float*)d_in[4];
  float* out = (float*)d_out;

  char* ws = (char*)d_ws;
  size_t off = 0;
  auto alloc = [&](size_t bytes) -> void* {
    void* p = ws + off;
    off += (bytes + 255) & ~(size_t)255;
    return p;
  };
  const size_t nX = (size_t)BS_ * D_;       // 8,388,608
  const size_t nW = (size_t)D_ * D_;        // 1,048,576
  u16*   Xbf  = (u16*)alloc(nX * 2);
  u16*   Wqb  = (u16*)alloc(nW * 2);
  u16*   Wkb  = (u16*)alloc(nW * 2);
  u16*   Wvb  = (u16*)alloc(nW * 2);
  u16*   Wob  = (u16*)alloc(nW * 2);
  float* RAW  = (float*)alloc(nX * 4);
  u16*   Qbf  = (u16*)alloc(nX * 2);
  u16*   Kbf  = (u16*)alloc(nX * 2);
  u16*   Vtb  = (u16*)alloc(nX * 2);
  u16*   Ctx  = (u16*)alloc(nX * 2);
  (void)ws_size; (void)in_sizes; (void)n_in; (void)out_size;

  // casts to bf16
  mha_cast_bf16<<<4096, 256, 0, stream>>>(x,  Xbf, (int)nX);
  mha_cast_bf16<<<1024, 256, 0, stream>>>(wq, Wqb, (int)nW);
  mha_cast_bf16<<<1024, 256, 0, stream>>>(wk, Wkb, (int)nW);
  mha_cast_bf16<<<1024, 256, 0, stream>>>(wv, Wvb, (int)nW);
  mha_cast_bf16<<<1024, 256, 0, stream>>>(wo, Wob, (int)nW);

  const int gemmBlocks = (BS_ / TM_) * (D_ / TN_);   // 64 * 16 = 1024 blocks

  // Q projection + RoPE
  mha_gemm_bf16<<<gemmBlocks, 256, 0, stream>>>(Xbf, Wqb, RAW, BS_, D_, D_);
  mha_rope_split<<<(BS_ * (D_ / 2)) / 256, 256, 0, stream>>>(RAW, Qbf, BS_ * (D_ / 2));
  // K projection + RoPE
  mha_gemm_bf16<<<gemmBlocks, 256, 0, stream>>>(Xbf, Wkb, RAW, BS_, D_, D_);
  mha_rope_split<<<(BS_ * (D_ / 2)) / 256, 256, 0, stream>>>(RAW, Kbf, BS_ * (D_ / 2));
  // V projection + transpose split
  mha_gemm_bf16<<<gemmBlocks, 256, 0, stream>>>(Xbf, Wvb, RAW, BS_, D_, D_);
  mha_v_split<<<(BS_ * D_) / 256, 256, 0, stream>>>(RAW, Vtb, BS_ * D_);

  // causal flash attention: 8192 waves = 1024 blocks x 8 waves
  mha_attention<<<1024, 256, 0, stream>>>(Qbf, Kbf, Vtb, Ctx);

  // output projection -> f32 d_out
  mha_gemm_bf16<<<gemmBlocks, 256, 0, stream>>>(Ctx, Wob, out, BS_, D_, D_);
}
